// AggregatedAttention_11845519802844
// MI455X (gfx1250) — matrix operand order repaired and checked by reading, not compiled, not run on hardware
//
#include <hip/hip_runtime.h>
#include <hip/hip_bf16.h>
#include <math.h>
#include <stdint.h>

// ---------------- types ----------------
typedef __attribute__((ext_vector_type(16))) __bf16 v16bf;
typedef __attribute__((ext_vector_type(8)))  float  v8f;

__device__ __forceinline__ unsigned short f32_to_bf16(float f) {
    unsigned int u = __float_as_uint(f);
    unsigned int r = u + 0x7FFFu + ((u >> 16) & 1u);   // round-to-nearest-even
    return (unsigned short)(r >> 16);
}

__device__ __forceinline__ float wave_sum(float v) {
    #pragma unroll
    for (int o = 16; o > 0; o >>= 1) v += __shfl_xor(v, o, 32);
    return v;
}
__device__ __forceinline__ float wave_max(float v) {
    #pragma unroll
    for (int o = 16; o > 0; o >>= 1) { float t = __shfl_xor(v, o, 32); v = t > v ? t : v; }
    return v;
}

// async global->LDS b128 copy (CDNA5, tracked by ASYNCcnt; no VGPR data round-trip)
__device__ __forceinline__ void async_g2l_b128(void* lds_dst, const void* gsrc) {
    unsigned lds = (unsigned)(uintptr_t)lds_dst;   // low 32 bits of flat addr = LDS offset
    asm volatile("global_load_async_to_lds_b128 %0, %1, off"
                 :: "v"(lds), "v"(gsrc) : "memory");
}
__device__ __forceinline__ void wait_async0() {
    asm volatile("s_wait_asynccnt 0x0" ::: "memory");
}

// ---------------- f32 -> bf16 convert ----------------
__global__ void k_f32_to_bf16(const float* __restrict__ src,
                              unsigned short* __restrict__ dst, int n) {
    int i = blockIdx.x * 256 + threadIdx.x;
    if (i < n) dst[i] = f32_to_bf16(src[i]);
}

// ---------------- WMMA bf16 GEMM: Y[M,N] = act(A[M,K] * W[N,K]^T + bias) ----
// block = 256 threads (8 waves); block tile 128(M) x 64(N).
// Each wave owns a 16(M) x 64(N) strip: 4 WMMAs per K-step, A fragment reused.
// Staging uses GLOBAL_LOAD_ASYNC_TO_LDS_B128 (ASYNCcnt), no VGPR round-trip.
// mode 0: none, 1: exact GELU.
__global__ __launch_bounds__(256) void k_gemm_bf16(
    const unsigned short* __restrict__ A,   // [M,K] bf16 row-major
    const unsigned short* __restrict__ Wt,  // [N,K] bf16 row-major
    const float* __restrict__ bias,         // [N]
    float* __restrict__ Y,                  // [M,N] f32
    int M, int N, int K, int mode)
{
    __shared__ unsigned int lAu[128 * 16];  // 128 rows x 32 bf16 (16 u32/row) = 8KB
    __shared__ unsigned int lBu[64 * 16];   // 64 n-rows x 32 k                = 4KB
    uint4* lA4 = reinterpret_cast<uint4*>(lAu);
    uint4* lB4 = reinterpret_cast<uint4*>(lBu);

    const int tid  = threadIdx.x;
    const int wave = tid >> 5;
    const int lane = tid & 31;
    const int bm   = blockIdx.x * 128;
    const int bn   = blockIdx.y * 64;
    const int m16  = lane & 15;
    const int half = lane >> 4;

    const uint4* Ag4 = reinterpret_cast<const uint4*>(A);
    const uint4* Bg4 = reinterpret_cast<const uint4*>(Wt);

    v8f acc0 = {}, acc1 = {}, acc2 = {}, acc3 = {};
    for (int k0 = 0; k0 < K; k0 += 32) {
        // A tile: 512 uint4 (128 rows x 4 uint4/row), 2 async b128 per thread
        #pragma unroll
        for (int r = 0; r < 2; ++r) {
            int idx = tid + r * 256;
            int row = idx >> 2, c4 = idx & 3;
            async_g2l_b128(&lA4[idx], Ag4 + ((size_t)(bm + row) * K + k0) / 8 + c4);
        }
        // B tile: 256 uint4 (64 rows x 4 uint4/row), 1 async b128 per thread
        {
            int row = tid >> 2, c4 = tid & 3;
            async_g2l_b128(&lB4[tid], Bg4 + ((size_t)(bn + row) * K + k0) / 8 + c4);
        }
        // prefetch next K tile into cache while WMMAs run on this one
        if (k0 + 32 < K) {
            __builtin_prefetch(Ag4 + ((size_t)(bm + (tid >> 1)) * K + k0 + 32) / 8, 0, 1);
            __builtin_prefetch(Bg4 + ((size_t)(bn + (tid & 63)) * K + k0 + 32) / 8, 0, 1);
        }
        wait_async0();          // this wave's async LDS writes are done
        __syncthreads();        // all waves' writes visible

        // A fragment: contiguous k order (k = 16*half + s), matching B
        union { v16bf v; unsigned int u[8]; } afr;
        #pragma unroll
        for (int i = 0; i < 8; ++i) {
            int ka  = half * 16 + i * 2;
            afr.u[i] = lAu[(wave * 16 + m16) * 16 + (ka >> 1)];
        }
        // 4 B fragments along N; reuse A across them
        #pragma unroll
        for (int nt = 0; nt < 4; ++nt) {
            union { v16bf v; unsigned int u[8]; } bfr;
            #pragma unroll
            for (int i = 0; i < 8; ++i) {
                int kb = half * 16 + i * 2;
                bfr.u[i] = lBu[(nt * 16 + m16) * 16 + (kb >> 1)];
            }
            v8f* accp = (nt == 0) ? &acc0 : (nt == 1) ? &acc1 : (nt == 2) ? &acc2 : &acc3;
            *accp = __builtin_amdgcn_wmma_f32_16x16x32_bf16(
                        false, afr.v, false, bfr.v, (short)0, *accp, false, false);
        }
        __syncthreads();
    }

    // epilogue: C layout -> row = (lane/16)*8 + r, col = lane%16
    #pragma unroll
    for (int nt = 0; nt < 4; ++nt) {
        const v8f acc = (nt == 0) ? acc0 : (nt == 1) ? acc1 : (nt == 2) ? acc2 : acc3;
        const int n = bn + nt * 16 + m16;
        const float bv = bias ? bias[n] : 0.0f;
        #pragma unroll
        for (int r = 0; r < 8; ++r) {
            int row = bm + wave * 16 + half * 8 + r;
            float v = acc[r] + bv;
            if (mode == 1) v = 0.5f * v * (1.0f + erff(v * 0.70710678118654752f));
            Y[(size_t)row * N + n] = v;
        }
    }
}

// ---------------- q-norm + q_s ----------------
// one block per token; wave = head (HD=32 == wave32)
__global__ __launch_bounds__(256) void k_qnorm(
    float* __restrict__ q, float* __restrict__ qs_out,
    const float* __restrict__ qemb, const float* __restrict__ temp,
    const float* __restrict__ sls)
{
    int tid = threadIdx.x; int h = tid >> 5; int d = tid & 31;
    size_t base = (size_t)blockIdx.x * 256 + tid;
    float v  = q[base];
    float ss = wave_sum(v * v);
    float qn = v / fmaxf(sqrtf(ss), 1e-12f);
    q[base]  = qn;                                // q_norm in place
    float sp = log1pf(__expf(temp[h]));           // softplus
    qs_out[base] = (qn + qemb[h * 32 + d]) * sp * sls[0];
}

// ---------------- per-head L2 normalize first 256 cols in place ----------------
__global__ __launch_bounds__(256) void k_headnorm(float* __restrict__ buf, int stride)
{
    int tid = threadIdx.x;
    size_t base = (size_t)blockIdx.x * stride + tid;
    float v  = buf[base];
    float ss = wave_sum(v * v);
    buf[base] = v / fmaxf(sqrtf(ss), 1e-12f);
}

// ---------------- 4x4 avg-pool + layernorm -> bf16 ----------------
__global__ __launch_bounds__(256) void k_pool_ln(
    const float* __restrict__ xs, const float* __restrict__ g,
    const float* __restrict__ bta, unsigned short* __restrict__ xpb)
{
    __shared__ float red[256];
    int c  = threadIdx.x;
    int bp = blockIdx.x; int b = bp >> 8; int pl = bp & 255;
    int ph = pl >> 4, pw = pl & 15;
    float s = 0.f;
    #pragma unroll
    for (int si = 0; si < 4; ++si)
        #pragma unroll
        for (int sj = 0; sj < 4; ++sj)
            s += xs[((size_t)(b * 4096 + (ph * 4 + si) * 64 + (pw * 4 + sj))) * 256 + c];
    float v = s * (1.0f / 16.0f);

    red[c] = v; __syncthreads();
    for (int st = 128; st > 0; st >>= 1) { if (c < st) red[c] += red[c + st]; __syncthreads(); }
    float mean = red[0] * (1.0f / 256.0f); __syncthreads();
    float dv = v - mean;
    red[c] = dv * dv; __syncthreads();
    for (int st = 128; st > 0; st >>= 1) { if (c < st) red[c] += red[c + st]; __syncthreads(); }
    float var = red[0] * (1.0f / 256.0f);
    float y = dv * rsqrtf(var + 1e-5f) * g[c] + bta[c];
    xpb[(size_t)bp * 256 + c] = f32_to_bf16(y);
}

// ---------------- CPB MLP: t2[l,h] = relu(table@w1^T+b1) @ w2^T + b2 ----------------
__global__ __launch_bounds__(256) void k_cpb(
    const float* __restrict__ table, const float* __restrict__ w1,
    const float* __restrict__ b1, const float* __restrict__ w2,
    const float* __restrict__ b2, float* __restrict__ t2)
{
    __shared__ float red[8][256];
    int t = threadIdx.x; int l = blockIdx.x;
    float x0 = table[l * 2 + 0], x1 = table[l * 2 + 1];
    float part[8] = {0, 0, 0, 0, 0, 0, 0, 0};
    for (int c = t; c < 512; c += 256) {
        float t1 = fmaxf(x0 * w1[c * 2 + 0] + x1 * w1[c * 2 + 1] + b1[c], 0.0f);
        #pragma unroll
        for (int h = 0; h < 8; ++h) part[h] += t1 * w2[h * 512 + c];
    }
    #pragma unroll
    for (int h = 0; h < 8; ++h) red[h][t] = part[h];
    __syncthreads();
    for (int st = 128; st > 0; st >>= 1) {
        if (t < st) {
            #pragma unroll
            for (int h = 0; h < 8; ++h) red[h][t] += red[h][t + st];
        }
        __syncthreads();
    }
    if (t < 8) t2[(size_t)l * 8 + t] = red[t][0] + b2[t];
}

// ---------------- fused attention: local(9) + pooled(256) softmax + context ----
// one block per (b,n); wave = head; lane = head-dim. Writes ctx in bf16 for proj GEMM.
__global__ __launch_bounds__(256) void k_attn(
    const float* __restrict__ qs,    // [B*N,256] q_s
    const float* __restrict__ qn,    // [B*N,256] q_norm
    const float* __restrict__ kvb,   // [B*N,512] cols 0..255 = normed k_l, 256..511 = v_l
    const float* __restrict__ kvp,   // [B*pL,512] cols 0..255 = normed k_p, 256..511 = v_p
    const float* __restrict__ t2,    // [L,8]
    const int*   __restrict__ rpi,   // [N*pL]
    const float* __restrict__ rpb,   // [8,9]
    const float* __restrict__ ltok,  // [8,32,9]
    const float* __restrict__ lbias, // [8,9]
    unsigned short* __restrict__ ctx)// [B*N,256] bf16
{
    __shared__ float qsh[8][32];
    __shared__ float pbuf[8][256];
    const int tid = threadIdx.x; const int h = tid >> 5; const int d = tid & 31;
    const int bn = blockIdx.x; const int b = bn >> 12; const int n = bn & 4095;
    const int i = n >> 6, j = n & 63;

    const size_t base = (size_t)bn * 256 + h * 32 + d;
    const float qsv = qs[base];
    const float qnv = qn[base];
    qsh[h][d] = qsv;
    __syncthreads();

    // ---- local scores over 3x3 window ----
    float aloc[9]; int nloc[9];
    float maxv = -1e30f;
    #pragma unroll
    for (int l = 0; l < 9; ++l) {
        int di = l / 3 - 1, dj = l % 3 - 1;
        int ii = i + di, jj = j + dj;
        bool oob = ((unsigned)ii >= 64u) || ((unsigned)jj >= 64u);
        int nl = ii * 64 + jj;
        nloc[l] = oob ? -1 : nl;
        float kvv = oob ? 0.f : kvb[((size_t)(b * 4096 + nl)) * 512 + h * 32 + d];
        float s = wave_sum(qsv * kvv) + rpb[h * 9 + l];
        aloc[l] = oob ? -1e30f : s;
        if (!oob && s > maxv) maxv = s;
    }

    // ---- pooled scores: lane owns 8 of 256 keys ----
    float sc[8];
    #pragma unroll
    for (int c = 0; c < 8; ++c) {
        int m = c * 32 + d;
        const float* kp = &kvp[((size_t)(b * 256 + m)) * 512 + h * 32];
        float s = 0.f;
        #pragma unroll
        for (int dd = 0; dd < 32; ++dd) s += qsh[h][dd] * kp[dd];
        int idx = rpi[n * 256 + m];
        s += t2[(size_t)idx * 8 + h];
        sc[c] = s;
        if (s > maxv) maxv = s;
    }
    maxv = wave_max(maxv);

    // ---- softmax over 9 + 256 ----
    float den = 0.f;
    #pragma unroll
    for (int c = 0; c < 8; ++c) { sc[c] = __expf(sc[c] - maxv); den += sc[c]; }
    den = wave_sum(den);
    #pragma unroll
    for (int l = 0; l < 9; ++l) { aloc[l] = __expf(aloc[l] - maxv); den += aloc[l]; }
    const float inv = 1.0f / den;
    #pragma unroll
    for (int c = 0; c < 8; ++c) pbuf[h][c * 32 + d] = sc[c] * inv;
    __syncthreads();

    // ---- local context: w_loc = q_norm . tokens + lbias + p_loc ----
    float xl = 0.f;
    #pragma unroll
    for (int l = 0; l < 9; ++l) {
        float wv = wave_sum(qnv * ltok[h * 288 + d * 9 + l]);
        wv += lbias[h * 9 + l] + aloc[l] * inv;
        float vv = (nloc[l] < 0) ? 0.f
                 : kvb[((size_t)(b * 4096 + nloc[l])) * 512 + 256 + h * 32 + d];
        xl += wv * vv;
    }

    // ---- pooled context: sum_m p[m] * v_p[m,d]  (coalesced across lanes) ----
    float xp = 0.f;
    for (int m = 0; m < 256; ++m)
        xp += pbuf[h][m] * kvp[((size_t)(b * 256 + m)) * 512 + 256 + h * 32 + d];

    ctx[base] = f32_to_bf16(xl + xp);
}

// ---------------- launcher ----------------
extern "C" void kernel_launch(void* const* d_in, const int* in_sizes, int n_in,
                              void* d_out, int out_size, void* d_ws, size_t ws_size,
                              hipStream_t stream) {
    (void)in_sizes; (void)n_in; (void)out_size; (void)ws_size;
    const float* x      = (const float*)d_in[0];
    const int*   rpi    = (const int*)  d_in[3];
    const float* rct    = (const float*)d_in[4];
    const float* sls    = (const float*)d_in[5];
    // d_in[6] = padding_mask (bool) -- recomputed analytically in-kernel
    const float* q_w    = (const float*)d_in[7];
    const float* q_b    = (const float*)d_in[8];
    const float* kv_w   = (const float*)d_in[9];
    const float* kv_b   = (const float*)d_in[10];
    const float* temp   = (const float*)d_in[11];
    const float* qemb   = (const float*)d_in[12];
    const float* sr_w   = (const float*)d_in[13];
    const float* sr_b   = (const float*)d_in[14];
    const float* ln_g   = (const float*)d_in[15];
    const float* ln_b   = (const float*)d_in[16];
    const float* cpb1_w = (const float*)d_in[17];
    const float* cpb1_b = (const float*)d_in[18];
    const float* cpb2_w = (const float*)d_in[19];
    const float* cpb2_b = (const float*)d_in[20];
    const float* rpb    = (const float*)d_in[21];
    const float* ltok   = (const float*)d_in[22];
    const float* lbias  = (const float*)d_in[23];
    const float* proj_w = (const float*)d_in[24];
    const float* proj_b = (const float*)d_in[25];

    const int BN = 2 * 4096;      // 8192 tokens
    const int BP = 2 * 256;       // 512 pooled tokens
    const int L  = 6241;

    char* ws = (char*)d_ws; size_t off = 0;
    auto salloc = [&](size_t bytes) { void* p = ws + off; off += (bytes + 255) & ~(size_t)255; return p; };
    unsigned short* xb   = (unsigned short*)salloc((size_t)BN * 256 * 2);
    unsigned short* wqb  = (unsigned short*)salloc((size_t)256 * 256 * 2);
    unsigned short* wkvb = (unsigned short*)salloc((size_t)512 * 256 * 2);
    unsigned short* wsrb = (unsigned short*)salloc((size_t)256 * 256 * 2);
    unsigned short* wprb = (unsigned short*)salloc((size_t)256 * 256 * 2);
    float* qbuf = (float*)salloc((size_t)BN * 256 * 4);   // raw q, becomes q_norm
    float* qsb  = (float*)salloc((size_t)BN * 256 * 4);   // q_s
    float* kvbf = (float*)salloc((size_t)BN * 512 * 4);   // kv (k normed in place)
    float* xsb  = (float*)salloc((size_t)BN * 256 * 4);   // gelu(sr)
    unsigned short* xpb = (unsigned short*)salloc((size_t)BP * 256 * 2);
    float* kvpb = (float*)salloc((size_t)BP * 512 * 4);   // kvp (k_p normed in place)
    float* t2b  = (float*)salloc((size_t)L * 8 * 4);
    unsigned short* ctxb = (unsigned short*)salloc((size_t)BN * 256 * 2);

    // f32 -> bf16 conversions
    k_f32_to_bf16<<<(BN * 256 + 255) / 256, 256, 0, stream>>>(x, xb, BN * 256);
    k_f32_to_bf16<<<(256 * 256 + 255) / 256, 256, 0, stream>>>(q_w, wqb, 256 * 256);
    k_f32_to_bf16<<<(512 * 256 + 255) / 256, 256, 0, stream>>>(kv_w, wkvb, 512 * 256);
    k_f32_to_bf16<<<(256 * 256 + 255) / 256, 256, 0, stream>>>(sr_w, wsrb, 256 * 256);
    k_f32_to_bf16<<<(256 * 256 + 255) / 256, 256, 0, stream>>>(proj_w, wprb, 256 * 256);

    // WMMA GEMMs: q, kv, gelu(sr)   (block tile 128x64)
    k_gemm_bf16<<<dim3(BN / 128, 256 / 64), 256, 0, stream>>>(xb, wqb,  q_b,  qbuf, BN, 256, 256, 0);
    k_gemm_bf16<<<dim3(BN / 128, 512 / 64), 256, 0, stream>>>(xb, wkvb, kv_b, kvbf, BN, 512, 256, 0);
    k_gemm_bf16<<<dim3(BN / 128, 256 / 64), 256, 0, stream>>>(xb, wsrb, sr_b, xsb,  BN, 256, 256, 1);

    k_qnorm<<<BN, 256, 0, stream>>>(qbuf, qsb, qemb, temp, sls);
    k_headnorm<<<BN, 256, 0, stream>>>(kvbf, 512);

    k_pool_ln<<<BP, 256, 0, stream>>>(xsb, ln_g, ln_b, xpb);
    k_gemm_bf16<<<dim3(BP / 128, 512 / 64), 256, 0, stream>>>(xpb, wkvb, kv_b, kvpb, BP, 512, 256, 0);
    k_headnorm<<<BP, 256, 0, stream>>>(kvpb, 512);

    k_cpb<<<L, 256, 0, stream>>>(rct, cpb1_w, cpb1_b, cpb2_w, cpb2_b, t2b);

    k_attn<<<BN, 256, 0, stream>>>(qsb, qbuf, kvbf, kvpb, t2b, rpi, rpb, ltok, lbias, ctxb);

    // final projection -> d_out (f32)
    k_gemm_bf16<<<dim3(BN / 128, 256 / 64), 256, 0, stream>>>(ctxb, wprb, proj_b, (float*)d_out, BN, 256, 256, 0);
}
